// DCN_Align_4320737100752
// MI455X (gfx1250) — matrix operand order repaired and checked by reading, not compiled, not run on hardware
//
#include <hip/hip_runtime.h>
#include <math.h>

// ---------- CDNA5 WMMA types ----------
typedef __attribute__((ext_vector_type(16))) __bf16 v16bf;
typedef __attribute__((ext_vector_type(8)))  float  v8f;
typedef __attribute__((ext_vector_type(8)))  unsigned short v8us;

#define BATCH 4
#define FH 128
#define FW 128

__device__ __forceinline__ unsigned short f2bf_bits(float f) {
    union { float f; unsigned u; } a; a.f = f;
    unsigned u = a.u + 0x7FFFu + ((a.u >> 16) & 1u);   // round-to-nearest-even
    return (unsigned short)(u >> 16);
}

__device__ __forceinline__ float lrelu_f(float v) { return v >= 0.f ? v : 0.1f * v; }

__device__ __forceinline__ float cubw(float t) {
    t = fabsf(t);
    const float a = -0.75f;
    if (t <= 1.f) return ((a + 2.f) * t - (a + 3.f)) * t * t + 1.f;
    if (t < 2.f)  return a * (((t - 5.f) * t + 8.f) * t - 4.f);
    return 0.f;
}

// ---------- weight repack: fp32 [M][Cin][kh][kw] -> bf16 [Mpad][K], K = Cin*kh*kw ----------
__global__ void pack_w_kernel(const float* __restrict__ w, unsigned short* __restrict__ dst,
                              int M, int Mpad, int K) {
    int n = Mpad * K;
    for (int i = blockIdx.x * 256 + threadIdx.x; i < n; i += gridDim.x * 256) {
        int m = i / K;
        int k = i - m * K;
        float v = (m < M) ? w[(size_t)m * K + k] : 0.f;
        dst[i] = f2bf_bits(v);
    }
}

// ---------- bicubic 2x upsample (align_corners=False, border clamp), a=-0.75 ----------
__global__ void upsample_bicubic_kernel(const float* __restrict__ in, float* __restrict__ out,
                                        int BC, int hin, int win) {
    const int hout = hin * 2, wout = win * 2;
    size_t total = (size_t)BC * hout * wout;
    size_t i = (size_t)blockIdx.x * 256 + threadIdx.x;
    if (i >= total) return;
    int x  = (int)(i % wout);
    int y  = (int)((i / wout) % hout);
    int bc = (int)(i / ((size_t)wout * hout));

    float fy = (y & 1) ? 0.25f : 0.75f;
    int   by = (y >> 1) - ((y & 1) ? 0 : 1);
    float fx = (x & 1) ? 0.25f : 0.75f;
    int   bx = (x >> 1) - ((x & 1) ? 0 : 1);

    float wy[4], wx[4];
    int   iy[4], ix[4];
#pragma unroll
    for (int j = 0; j < 4; ++j) {
        wy[j] = cubw(fy - (float)(j - 1));
        wx[j] = cubw(fx - (float)(j - 1));
        int ty = by + (j - 1); iy[j] = ty < 0 ? 0 : (ty > hin - 1 ? hin - 1 : ty);
        int tx = bx + (j - 1); ix[j] = tx < 0 ? 0 : (tx > win - 1 ? win - 1 : tx);
    }
    const float* src = in + (size_t)bc * hin * win;
    float acc = 0.f;
#pragma unroll
    for (int j = 0; j < 4; ++j) {
        float rowv = 0.f;
#pragma unroll
        for (int k = 0; k < 4; ++k) rowv += wx[k] * src[iy[j] * win + ix[k]];
        acc += wy[j] * rowv;
    }
    out[i] = acc;
}

// ---------- implicit-GEMM conv via WMMA bf16, 2x2 register tiling ----------
// Workgroup: 4 wave32 (128 thr) -> 64 outch x 64 pixels; each wave owns a
// 32Mx32N tile = 2x2 WMMA tiles, so each K-step loads 2 A + 2 B fragments and
// issues 4 v_wmma (fragment reuse halves memory ops per WMMA).
// im2col staged in LDS as [col][Kchunk] bf16 (contiguous 32-byte B fragments).
// K chunk = 288 (KS=3, lcm(32,9) -> whole channels) keeps LDS at 37.9 KB.
template<int KS>
__global__ __launch_bounds__(128) void conv_wmma_kernel(
    const float* __restrict__ inA, int CinA,
    const float* __restrict__ inB, int CinB, float scaleB,
    const unsigned short* __restrict__ Wp,   // bf16 bits, [Mpad][Ktot]
    const float* __restrict__ bias,
    float* __restrict__ out, int Cout,
    int Hs, int Wd, int act)
{
    constexpr int PAD  = KS / 2;
    constexpr int TAPS = KS * KS;
    constexpr int KCH  = (KS == 3) ? 288 : 128;   // K elems per LDS chunk
    constexpr int CCH  = KCH / TAPS;              // channels per chunk (32 / 128)
    constexpr int BST  = KCH + 8;                 // row stride (u16), mult of 8
    __shared__ unsigned short bmat[64 * BST];     // 37,888 B (KS=3) / 17,408 B (KS=1)

    const int tid = threadIdx.x;
    const int tpr = Wd / 64;
    const int t   = blockIdx.x;
    const int x0  = (t % tpr) * 64;
    const int row = t / tpr;
    const int y   = row % Hs;
    const int b   = row / Hs;
    const int Cin = CinA + CinB;
    const int Ktot = Cin * TAPS;

    const int col = tid & 63;     // pixel within tile (staging)
    const int grp = tid >> 6;     // 0..1

    const int lane = tid & 31;
    const int wave = tid >> 5;    // 0..3
    const int mt2  = wave & 1;    // M half (32 rows)
    const int nt2  = wave >> 1;   // N half (32 cols)
    const int n16  = lane & 15;
    const int g16  = lane >> 4;

    const unsigned short* abase0 =
        Wp + (size_t)(blockIdx.y * 64 + mt2 * 32 + n16) * Ktot + g16 * 16;
    const unsigned short* abase1 = abase0 + (size_t)16 * Ktot;
    const unsigned short* bbase0 = bmat + (nt2 * 32 + n16) * BST + g16 * 16;
    const unsigned short* bbase1 = bbase0 + 16 * BST;

    v8f acc00 = {0.f,0.f,0.f,0.f,0.f,0.f,0.f,0.f};
    v8f acc01 = acc00, acc10 = acc00, acc11 = acc00;

    for (int k0 = 0; k0 < Ktot; k0 += KCH) {
        // ---- stage im2col chunk: channels [c0, c0+CCH) for 64 pixels ----
        const int c0 = k0 / TAPS;
        for (int c = c0 + grp; c < c0 + CCH; c += 2) {
            const float* src;
            int ch; float sc;
            if (c < CinA) { src = inA; ch = c;        sc = 1.f; }
            else          { src = inB; ch = c - CinA; sc = scaleB; }
            const int csel = (c < CinA) ? CinA : CinB;
            const float* base = src + (((size_t)b * csel + ch) * Hs) * Wd;
            unsigned short* drow = bmat + col * BST + (c * TAPS - k0);
#pragma unroll
            for (int tap = 0; tap < TAPS; ++tap) {
                const int ry = tap / KS;
                const int cx = tap - ry * KS;
                const int gy = y - PAD + ry;
                const int gx = x0 + col - PAD + cx;
                float v = (gy >= 0 && gy < Hs && gx >= 0 && gx < Wd)
                            ? base[gy * Wd + gx] * sc : 0.f;
                drow[tap] = f2bf_bits(v);
            }
        }
        __syncthreads();

        // ---- GEMM over this chunk: 4 WMMAs per K-step (2x2 reuse) ----
        for (int kt = 0; kt < KCH / 32; ++kt) {
            const int ko = kt * 32;
            __builtin_prefetch((const void*)(abase0 + k0 + ko + 128), 0, 0);
            union { v16bf v; v8us h[2]; } a0, a1, b0, b1;
            const v8us* ap0 = (const v8us*)(abase0 + k0 + ko);
            const v8us* ap1 = (const v8us*)(abase1 + k0 + ko);
            a0.h[0] = ap0[0]; a0.h[1] = ap0[1];
            a1.h[0] = ap1[0]; a1.h[1] = ap1[1];
            const v8us* bp0 = (const v8us*)(bbase0 + ko);
            const v8us* bp1 = (const v8us*)(bbase1 + ko);
            b0.h[0] = bp0[0]; b0.h[1] = bp0[1];
            b1.h[0] = bp1[0]; b1.h[1] = bp1[1];
            acc00 = __builtin_amdgcn_wmma_f32_16x16x32_bf16(
                false, a0.v, false, b0.v, (short)0, acc00, false, false);
            acc01 = __builtin_amdgcn_wmma_f32_16x16x32_bf16(
                false, a0.v, false, b1.v, (short)0, acc01, false, false);
            acc10 = __builtin_amdgcn_wmma_f32_16x16x32_bf16(
                false, a1.v, false, b0.v, (short)0, acc10, false, false);
            acc11 = __builtin_amdgcn_wmma_f32_16x16x32_bf16(
                false, a1.v, false, b1.v, (short)0, acc11, false, false);
        }
        __syncthreads();
    }

    // ---- epilogue: bias + activation + store 2x2 tiles ----
#pragma unroll
    for (int tm = 0; tm < 2; ++tm) {
#pragma unroll
        for (int tn = 0; tn < 2; ++tn) {
            const v8f& a = tm == 0 ? (tn == 0 ? acc00 : acc01)
                                   : (tn == 0 ? acc10 : acc11);
            const int xo = x0 + nt2 * 32 + tn * 16 + n16;
#pragma unroll
            for (int r = 0; r < 8; ++r) {
                int co = blockIdx.y * 64 + mt2 * 32 + tm * 16 + g16 * 8 + r;
                if (co < Cout) {
                    float v = a[r] + bias[co];
                    if (act) v = lrelu_f(v);
                    out[(((size_t)b * Cout + co) * Hs + y) * Wd + xo] = v;
                }
            }
        }
    }
}

// ---------- fused modulated deformable conv (DCNv2): gather->LDS->WMMA ----------
// om: [B,216,H,W] = (dy[72] | dx[72] | mask[72]), weights packed [64][576] bf16.
// Sample buffer stored [pixel][K] so phase-2 B fragments are contiguous 32-byte loads.
__global__ __launch_bounds__(256) void dcn_wmma_kernel(
    const float* __restrict__ x,
    const float* __restrict__ om,
    const unsigned short* __restrict__ Wp,
    const float* __restrict__ bias,
    float* __restrict__ out, int act)
{
    constexpr int BST = 576 + 8;                  // 584, mult of 8
    __shared__ unsigned short vbuf[32 * BST];     // 37,376 B
    const int tid = threadIdx.x;
    const int t   = blockIdx.x;
    const int tpr = FW / 32;
    const int x0  = (t % tpr) * 32;
    const int row = t / tpr;
    const int y   = row % FH;
    const int b   = row / FH;
    const size_t HW = (size_t)FH * FW;

    // phase 1: bilinear gather + modulation -> bf16 B-matrix [N=32][K=576] in LDS
    for (int task = tid; task < 72 * 32; task += 256) {
        int gk = task >> 5;
        int n  = task & 31;
        int g  = gk / 9;
        int k  = gk - g * 9;
        int xp = x0 + n;
        size_t obase = (((size_t)b * 216) * FH + y) * FW + xp;
        float dy = om[obase + (size_t)(g * 9 + k) * HW];
        float dx = om[obase + (size_t)(72 + g * 9 + k) * HW];
        float mm = om[obase + (size_t)(144 + g * 9 + k) * HW];
        mm = 1.f / (1.f + __expf(-mm));
        float py = (float)y  + (float)(k / 3 - 1) + dy;
        float px = (float)xp + (float)(k % 3 - 1) + dx;
        float fy = floorf(py), fx = floorf(px);
        float wy = py - fy,    wx = px - fx;
        int iy0 = (int)fy, ix0 = (int)fx;
        int iy1 = iy0 + 1, ix1 = ix0 + 1;
        float vy0 = (iy0 >= 0 && iy0 < FH) ? 1.f : 0.f;
        float vy1 = (iy1 >= 0 && iy1 < FH) ? 1.f : 0.f;
        float vx0 = (ix0 >= 0 && ix0 < FW) ? 1.f : 0.f;
        float vx1 = (ix1 >= 0 && ix1 < FW) ? 1.f : 0.f;
        int cy0 = min(max(iy0, 0), FH - 1);
        int cy1 = min(max(iy1, 0), FH - 1);
        int cx0 = min(max(ix0, 0), FW - 1);
        int cx1 = min(max(ix1, 0), FW - 1);
        float w00 = (1.f - wy) * (1.f - wx) * vy0 * vx0 * mm;
        float w01 = (1.f - wy) * wx         * vy0 * vx1 * mm;
        float w10 = wy         * (1.f - wx) * vy1 * vx0 * mm;
        float w11 = wy         * wx         * vy1 * vx1 * mm;
        const float* xb = x + (((size_t)b * 64 + g * 8) * FH) * FW;
        unsigned short* drow = vbuf + n * BST;
#pragma unroll
        for (int c = 0; c < 8; ++c) {
            const float* ch = xb + (size_t)c * HW;
            float val = w00 * ch[cy0 * FW + cx0] + w01 * ch[cy0 * FW + cx1]
                      + w10 * ch[cy1 * FW + cx0] + w11 * ch[cy1 * FW + cx1];
            int kk = (g * 8 + c) * 9 + k;       // matches weight reshape (o,g,c,k)
            drow[kk] = f2bf_bits(val);
        }
    }
    __syncthreads();

    // phase 2: GEMM  out[64 x 32] = W[64 x 576] * V[576 x 32]
    const int lane = tid & 31;
    const int wave = tid >> 5;
    const int mt   = wave & 3;
    const int nt   = wave >> 2;
    const int n16  = lane & 15;
    const int g16  = lane >> 4;
    const unsigned short* abase = Wp + (size_t)(mt * 16 + n16) * 576 + g16 * 16;
    const unsigned short* bbase = vbuf + (nt * 16 + n16) * BST + g16 * 16;

    v8f acc = {0.f, 0.f, 0.f, 0.f, 0.f, 0.f, 0.f, 0.f};
#pragma unroll 2
    for (int kt = 0; kt < 18; ++kt) {
        __builtin_prefetch((const void*)(abase + kt * 32 + 128), 0, 0);
        union { v16bf v; v8us h[2]; } av, bv;
        const v8us* ap = (const v8us*)(abase + kt * 32);
        av.h[0] = ap[0];
        av.h[1] = ap[1];
        const v8us* bp = (const v8us*)(bbase + kt * 32);
        bv.h[0] = bp[0];
        bv.h[1] = bp[1];
        acc = __builtin_amdgcn_wmma_f32_16x16x32_bf16(
            false, av.v, false, bv.v, (short)0, acc, false, false);
    }

#pragma unroll
    for (int r = 0; r < 8; ++r) {
        int co = mt * 16 + g16 * 8 + r;
        float v = acc[r] + bias[co];
        if (act) v = lrelu_f(v);
        out[(((size_t)b * 64 + co) * FH + y) * FW + (x0 + nt * 16 + n16)] = v;
    }
}

// ---------- orchestration ----------
extern "C" void kernel_launch(void* const* d_in, const int* in_sizes, int n_in,
                              void* d_out, int out_size, void* d_ws, size_t ws_size,
                              hipStream_t stream) {
    (void)in_sizes; (void)n_in; (void)out_size; (void)ws_size;

    const float* nbr  = (const float*)d_in[0];
    const float* ref  = (const float*)d_in[1];
    const float* lowf = (const float*)d_in[2];
    const float* lowo = (const float*)d_in[3];
    const float* w_oc1  = (const float*)d_in[4];  const float* b_oc1  = (const float*)d_in[5];
    const float* w_oc2  = (const float*)d_in[6];  const float* b_oc2  = (const float*)d_in[7];
    const float* w_oc3  = (const float*)d_in[8];  const float* b_oc3  = (const float*)d_in[9];
    const float* w_lo   = (const float*)d_in[10]; const float* b_lo   = (const float*)d_in[11];
    const float* w_lf   = (const float*)d_in[12]; const float* b_lf   = (const float*)d_in[13];
    const float* w_fea  = (const float*)d_in[14]; const float* b_fea  = (const float*)d_in[15];
    const float* w_cas1 = (const float*)d_in[16]; const float* b_cas1 = (const float*)d_in[17];
    const float* w_cas2 = (const float*)d_in[18]; const float* b_cas2 = (const float*)d_in[19];
    const float* w_om1  = (const float*)d_in[20]; const float* b_om1  = (const float*)d_in[21];
    const float* w_d1   = (const float*)d_in[22]; const float* b_d1   = (const float*)d_in[23];
    const float* w_om2  = (const float*)d_in[24]; const float* b_om2  = (const float*)d_in[25];
    const float* w_d2   = (const float*)d_in[26]; const float* b_d2   = (const float*)d_in[27];

    float* out = (float*)d_out;
    float* ws  = (float*)d_ws;
    const size_t F = (size_t)BATCH * 64 * FH * FW;      // 4,194,304 floats per 64-ch map

    float* t_off  = ws;
    float* t_tmp  = ws + F;
    float* t_up   = ws + 2 * F;
    float* t_al   = ws + 3 * F;
    float* t_om   = ws + 4 * F;                          // 216 ch full-res
    float* t_half = t_om + (size_t)BATCH * 216 * FH * FW;
    unsigned short* wp = (unsigned short*)(t_half + (size_t)BATCH * 64 * 64 * 64);

    size_t wo = 0;
    auto nxt = [&](size_t n) { unsigned short* r = wp + wo; wo += n; return r; };
    unsigned short* p_oc1  = nxt((size_t)64 * 1152);
    unsigned short* p_oc2  = nxt((size_t)64 * 1152);
    unsigned short* p_oc3  = nxt((size_t)64 * 576);
    unsigned short* p_lo   = nxt((size_t)64 * 128);
    unsigned short* p_lf   = nxt((size_t)64 * 128);
    unsigned short* p_fea  = nxt((size_t)64 * 1152);
    unsigned short* p_cas1 = nxt((size_t)64 * 1152);
    unsigned short* p_cas2 = nxt((size_t)64 * 576);
    unsigned short* p_om1  = nxt((size_t)256 * 576);
    unsigned short* p_d1   = nxt((size_t)64 * 576);
    unsigned short* p_om2  = nxt((size_t)256 * 576);
    unsigned short* p_d2   = nxt((size_t)64 * 576);

    auto pack = [&](const float* w, unsigned short* dst, int M, int Mpad, int K) {
        int n = Mpad * K;
        pack_w_kernel<<<dim3((n + 255) / 256), dim3(256), 0, stream>>>(w, dst, M, Mpad, K);
    };
    pack(w_oc1,  p_oc1,  64,  64,  1152);
    pack(w_oc2,  p_oc2,  64,  64,  1152);
    pack(w_oc3,  p_oc3,  64,  64,  576);
    pack(w_lo,   p_lo,   64,  64,  128);
    pack(w_lf,   p_lf,   64,  64,  128);
    pack(w_fea,  p_fea,  64,  64,  1152);
    pack(w_cas1, p_cas1, 64,  64,  1152);
    pack(w_cas2, p_cas2, 64,  64,  576);
    pack(w_om1,  p_om1,  216, 256, 576);
    pack(w_d1,   p_d1,   64,  64,  576);
    pack(w_om2,  p_om2,  216, 256, 576);
    pack(w_d2,   p_d2,   64,  64,  576);

    auto conv3 = [&](const float* inA, int CinA, const float* inB, int CinB, float sB,
                     const unsigned short* Wq, const float* bias, float* op,
                     int Cout, int Mpad, int act) {
        dim3 g(BATCH * FH * (FW / 64), Mpad / 64);
        conv_wmma_kernel<3><<<g, dim3(128), 0, stream>>>(
            inA, CinA, inB, CinB, sB, Wq, bias, op, Cout, FH, FW, act);
    };
    auto conv1h = [&](const float* inA, const unsigned short* Wq, const float* bias, float* op) {
        dim3 g(BATCH * 64 * (64 / 64), 1);
        conv_wmma_kernel<1><<<g, dim3(128), 0, stream>>>(
            inA, 128, (const float*)nullptr, 0, 1.f, Wq, bias, op, 64, 64, 64, 0);
    };
    auto upsamp = [&](const float* in, float* op) {
        size_t n = (size_t)BATCH * 64 * FH * FW;
        upsample_bicubic_kernel<<<dim3((unsigned)((n + 255) / 256)), dim3(256), 0, stream>>>(
            in, op, BATCH * 64, 64, 64);
    };

    // off = lrelu(conv(cat(nbr, ref), oc1))
    conv3(nbr, 64, ref, 64, 1.f, p_oc1, b_oc1, t_off, 64, 64, 1);
    // lo = upsample(conv1x1(low_offset)); concat scale 2.0 folded into next conv
    conv1h(lowo, p_lo, b_lo, t_half);
    upsamp(t_half, t_up);
    // off = lrelu(conv(cat(off, lo*2), oc2)); off = lrelu(conv(off, oc3))
    conv3(t_off, 64, t_up, 64, 2.f, p_oc2, b_oc2, t_tmp, 64, 64, 1);
    conv3(t_tmp, 64, nullptr, 0, 1.f, p_oc3, b_oc3, t_off, 64, 64, 1);
    // om1 = conv(off); aligned_raw = dcn(nbr, om1)
    conv3(t_off, 64, nullptr, 0, 1.f, p_om1, b_om1, t_om, 216, 256, 0);
    dcn_wmma_kernel<<<dim3(BATCH * FH * (FW / 32)), dim3(256), 0, stream>>>(
        nbr, t_om, p_d1, b_d1, t_tmp, 0);
    // lf = upsample(conv1x1(low_fea)); aligned = lrelu(conv(cat(aligned_raw, lf), fea))
    conv1h(lowf, p_lf, b_lf, t_half);
    upsamp(t_half, t_up);
    conv3(t_tmp, 64, t_up, 64, 1.f, p_fea, b_fea, t_al, 64, 64, 1);
    // off = lrelu(conv(cat(aligned, ref), cas1)); off = lrelu(conv(off, cas2)) -> output 2
    conv3(t_al, 64, ref, 64, 1.f, p_cas1, b_cas1, t_off, 64, 64, 1);
    float* out_off = out + F;
    conv3(t_off, 64, nullptr, 0, 1.f, p_cas2, b_cas2, out_off, 64, 64, 1);
    // om2 = conv(off); aligned = lrelu(dcn(aligned, om2)) -> output 1
    conv3(out_off, 64, nullptr, 0, 1.f, p_om2, b_om2, t_om, 216, 256, 0);
    dcn_wmma_kernel<<<dim3(BATCH * FH * (FW / 32)), dim3(256), 0, stream>>>(
        t_al, t_om, p_d2, b_d2, out, 1);
}
